// BiLSTM_CRF_MODIFY_PARALLEL_15436112462113
// MI455X (gfx1250) — compile-verified, matrix operand order checked by reading, not run on hardware
//
#include <hip/hip_runtime.h>
#include <hip/hip_bf16.h>
#include <math.h>

// ---------------------------------------------------------------------------
// BiLSTM-CRF for MI455X (gfx1250).
//   SEQ=8192, EMB=512, H=512/dir (4H=2048 gates), TAGSET=9.
// Pipeline (all on `stream`):
//   1) gather emb -> x (bf16)  +  convert Wih cat (bf16)  +  pack-transpose Whh
//   2) WMMA GEMM (software-pipelined): xproj[8192,4096] = x @ [Wih_f;Wih_b]^T + bias
//   3) sequential LSTM recurrence, 1 workgroup/direction (grid=2), h in LDS,
//      WhhT streamed from L2 as packed bf16 quads (4 fma per 64-bit load)
//   4) feats = concat(h_f,h_b) @ W_out^T + b_out
//   5) Viterbi forward scan (single wave) + serial backtrace
// ---------------------------------------------------------------------------

#define SEQ   8192
#define EMBD  512
#define HDIR  512      // hidden per direction
#define G4H   2048     // 4*HDIR gates per direction
#define NCAT  4096     // both directions' gates
#define NTAG  9
#define START_IDX 7
#define STOP_IDX  8

typedef __attribute__((ext_vector_type(16))) __bf16   v16bf;
typedef __attribute__((ext_vector_type(8)))  float    v8f;
typedef __attribute__((ext_vector_type(4)))  unsigned int u32x4;

union Frag16 { u32x4 u[2]; v16bf v; };

__device__ __forceinline__ unsigned short f2bf(float f) {
  unsigned int u = __float_as_uint(f);
  unsigned int r = u + 0x7FFFu + ((u >> 16) & 1u);   // round-to-nearest-even
  return (unsigned short)(r >> 16);
}
__device__ __forceinline__ float bf2f(unsigned int lo16) {
  return __uint_as_float(lo16 << 16);
}
__device__ __forceinline__ float sigm(float x) {
  return 1.0f / (1.0f + __expf(-x));
}

// ---- 1a: embedding gather -> bf16 x [SEQ, EMB] ----------------------------
__global__ void k_gather_bf16(const int* __restrict__ sent,
                              const float* __restrict__ emb,
                              unsigned short* __restrict__ xbf) {
  int i = blockIdx.x * blockDim.x + threadIdx.x;        // SEQ*EMB
  int t = i >> 9, e = i & 511;
  xbf[i] = f2bf(emb[(long)sent[t] * EMBD + e]);
}

// ---- 1b: Wih concat (f then b) -> bf16 [NCAT, EMB] ------------------------
__global__ void k_cvt_wih(const float* __restrict__ wf,
                          const float* __restrict__ wb,
                          unsigned short* __restrict__ out) {
  int i = blockIdx.x * blockDim.x + threadIdx.x;        // NCAT*EMB
  int n = i >> 9, k = i & 511;
  float v = (n < G4H) ? wf[n * EMBD + k] : wb[(n - G4H) * EMBD + k];
  out[i] = f2bf(v);
}

// ---- 1c: Whh[4H,H] -> packed WhhT bf16 ------------------------------------
// Layout: for k2 in [0,256), pair p in [0,1024): one uint2 holding
//   { bf16 W[2p][2k2], bf16 W[2p+1][2k2], bf16 W[2p][2k2+1], bf16 W[2p+1][2k2+1] }
// flat ushort index = k2*4096 + p*4 + (k&1)*2 + (g&1)
__global__ void k_trans_whh_pack(const float* __restrict__ w,
                                 unsigned short* __restrict__ out) {
  int i = blockIdx.x * blockDim.x + threadIdx.x;        // HDIR*G4H, i=k*G4H+g
  int k = i >> 11, g = i & 2047;
  size_t idx = (size_t)(k >> 1) * 4096 + (size_t)(g >> 1) * 4 + (k & 1) * 2 + (g & 1);
  out[idx] = f2bf(w[g * HDIR + k]);
}

// ---- 2: WMMA GEMM  C[SEQ,NCAT] = A[SEQ,EMB] * W[NCAT,EMB]^T + bias --------
// Each wave: one 16x64 strip (A fragment reused over 4 B fragments).
// Software-pipelined: next k-chunk's fragments are loaded before the current
// chunk's 4 WMMAs, so loads stay one iteration ahead of their consumers.
__global__ __launch_bounds__(256)
void k_gemm_xproj(const unsigned short* __restrict__ A,
                  const unsigned short* __restrict__ W,
                  const float* __restrict__ bih_f,
                  const float* __restrict__ bih_b,
                  float* __restrict__ C) {
  const int lane = threadIdx.x & 31;
  const int wid  = (blockIdx.x * blockDim.x + threadIdx.x) >> 5;
  const int m0 = (wid >> 6) << 4;     // 512 row-tiles of 16
  const int n0 = (wid & 63) << 6;     // 64 col-strips of 64
  const int l  = lane & 15;
  const int hi = lane >> 4;           // selects K half per ISA 16-bit layout

  v8f acc[4] = {};
  const unsigned short* arow  = A + (long)(m0 + l) * EMBD + hi * 8;
  const unsigned short* bbase = W + (long)(n0 + l) * EMBD + hi * 8;

  Frag16 fa, fb[4];
  fa.u[0] = *(const u32x4*)(arow);
  fa.u[1] = *(const u32x4*)(arow + 16);
#pragma unroll
  for (int j = 0; j < 4; ++j) {
    const unsigned short* br = bbase + (long)j * 16 * EMBD;
    fb[j].u[0] = *(const u32x4*)(br);
    fb[j].u[1] = *(const u32x4*)(br + 16);
  }

#pragma unroll
  for (int k0 = 0; k0 < EMBD - 32; k0 += 32) {
    const int kn = k0 + 32;
    Frag16 na, nb[4];
    na.u[0] = *(const u32x4*)(arow + kn);
    na.u[1] = *(const u32x4*)(arow + kn + 16);
#pragma unroll
    for (int j = 0; j < 4; ++j) {
      const unsigned short* br = bbase + (long)j * 16 * EMBD + kn;
      nb[j].u[0] = *(const u32x4*)(br);
      nb[j].u[1] = *(const u32x4*)(br + 16);
    }
#pragma unroll
    for (int j = 0; j < 4; ++j)
      acc[j] = __builtin_amdgcn_wmma_f32_16x16x32_bf16(
          false, fa.v, false, fb[j].v, (short)0, acc[j], false, false);
    fa = na;
#pragma unroll
    for (int j = 0; j < 4; ++j) fb[j] = nb[j];
  }
#pragma unroll
  for (int j = 0; j < 4; ++j)   // epilogue: last k-chunk
    acc[j] = __builtin_amdgcn_wmma_f32_16x16x32_bf16(
        false, fa.v, false, fb[j].v, (short)0, acc[j], false, false);

#pragma unroll
  for (int j = 0; j < 4; ++j) {
    int ncol = n0 + j * 16 + l;
    float bias = (ncol < G4H) ? bih_f[ncol] : bih_b[ncol - G4H];
#pragma unroll
    for (int r = 0; r < 8; ++r) {
      int row = m0 + r + hi * 8;
      C[(long)row * NCAT + ncol] = acc[j][r] + bias;
    }
  }
}

// ---- 3: sequential LSTM recurrence, one workgroup per direction -----------
__global__ __launch_bounds__(1024)
void k_lstm_rec(const float* __restrict__ xproj,           // [SEQ, NCAT]
                const unsigned short* __restrict__ whhT_f, // packed, see 1c
                const unsigned short* __restrict__ whhT_b,
                const float* __restrict__ bhh_f,
                const float* __restrict__ bhh_b,
                float* __restrict__ h_f,                   // [SEQ, HDIR]
                float* __restrict__ h_b) {
  __shared__ float sh[HDIR];
  __shared__ float sg[G4H];
  const int dir = blockIdx.x;
  const uint2* wp2 = (const uint2*)(const void*)(dir ? whhT_b : whhT_f);
  const float* bhh = dir ? bhh_b : bhh_f;
  float* hout = dir ? h_b : h_f;
  const float* xp = xproj + dir * G4H;
  const int tid = threadIdx.x;
  const int g0 = tid * 2;
  wp2 += tid;                                    // pair index = tid

  if (tid < HDIR) sh[tid] = 0.0f;
  float c = 0.0f;
  const float b0 = bhh[g0], b1 = bhh[g0 + 1];
  __syncthreads();

  for (int step = 0; step < SEQ; ++step) {
    const int t = dir ? (SEQ - 1 - step) : step;
    const float* xrow = xp + (long)t * NCAT;
    float a0 = xrow[g0] + b0;
    float a1 = xrow[g0 + 1] + b1;
#pragma unroll 4
    for (int k2 = 0; k2 < HDIR / 2; ++k2) {
      uint2 w = wp2[k2 * 1024];                  // 4 bf16: (k,k+1)x(g0,g1)
      float h0 = sh[2 * k2];
      float h1 = sh[2 * k2 + 1];
      a0 = fmaf(h0, bf2f(w.x & 0xffffu), a0);
      a1 = fmaf(h0, bf2f(w.x >> 16), a1);
      a0 = fmaf(h1, bf2f(w.y & 0xffffu), a0);
      a1 = fmaf(h1, bf2f(w.y >> 16), a1);
    }
    sg[g0] = a0;
    sg[g0 + 1] = a1;
    __syncthreads();
    if (tid < HDIR) {
      float ig = sigm(sg[tid]);
      float fg = sigm(sg[HDIR + tid]);
      float gg = tanhf(sg[2 * HDIR + tid]);
      float og = sigm(sg[3 * HDIR + tid]);
      c = fg * c + ig * gg;
      float h = og * tanhf(c);
      sh[tid] = h;
      hout[(long)t * HDIR + tid] = h;
    }
    __syncthreads();
  }
}

// ---- 4: feats[t,9] = concat(h_f,h_b)[t] . W_out^T + b_out -----------------
__global__ __launch_bounds__(256)
void k_feats(const float* __restrict__ hf, const float* __restrict__ hb,
             const float* __restrict__ Wout,  // [NTAG, 2*HDIR]
             const float* __restrict__ bout,
             float* __restrict__ feats) {
  const int t = blockIdx.x;
  const int tid = threadIdx.x;
  float acc[NTAG];
#pragma unroll
  for (int j = 0; j < NTAG; ++j) acc[j] = 0.0f;
  for (int h = tid; h < 2 * HDIR; h += 256) {
    float v = (h < HDIR) ? hf[(long)t * HDIR + h] : hb[(long)t * HDIR + h - HDIR];
#pragma unroll
    for (int j = 0; j < NTAG; ++j) acc[j] += v * Wout[j * 2 * HDIR + h];
  }
  __shared__ float red[256 * NTAG];
#pragma unroll
  for (int j = 0; j < NTAG; ++j) red[tid * NTAG + j] = acc[j];
  for (int s = 128; s > 0; s >>= 1) {
    __syncthreads();
    if (tid < s)
#pragma unroll
      for (int j = 0; j < NTAG; ++j)
        red[tid * NTAG + j] += red[(tid + s) * NTAG + j];
  }
  if (tid == 0)
#pragma unroll
    for (int j = 0; j < NTAG; ++j) feats[(long)t * NTAG + j] = red[j] + bout[j];
}

// ---- 5: Viterbi: forward scan + backtrace ---------------------------------
__global__ void k_viterbi(const float* __restrict__ feats,
                          const float* __restrict__ trans,  // [next, prev]
                          int* __restrict__ bp,             // [SEQ, NTAG]
                          float* __restrict__ out) {        // [1 + SEQ]
  __shared__ float fv[16];
  __shared__ float fvn[16];
  const int tid = threadIdx.x;
  if (tid < NTAG) fv[tid] = (tid == START_IDX) ? 0.0f : -10000.0f;
  __syncthreads();
  for (int t = 0; t < SEQ; ++t) {
    if (tid < NTAG) {
      float best = -3.4e38f;
      int arg = 0;
#pragma unroll
      for (int p = 0; p < NTAG; ++p) {
        float s = fv[p] + trans[tid * NTAG + p];
        if (s > best) { best = s; arg = p; }
      }
      fvn[tid] = best + feats[(long)t * NTAG + tid];
      bp[(long)t * NTAG + tid] = arg;
    }
    __syncthreads();
    if (tid < NTAG) fv[tid] = fvn[tid];
    __syncthreads();
  }
  if (tid == 0) {
    float best = -3.4e38f;
    int arg = 0;
#pragma unroll
    for (int p = 0; p < NTAG; ++p) {
      float s = fv[p] + trans[STOP_IDX * NTAG + p];
      if (s > best) { best = s; arg = p; }
    }
    out[0] = best;                       // path score
    int c = arg;
    for (int t = SEQ - 1; t >= 0; --t) { // path[T-1] = argmax(terminal)
      out[1 + t] = (float)c;
      c = bp[(long)t * NTAG + c];
    }
  }
}

// ---------------------------------------------------------------------------
extern "C" void kernel_launch(void* const* d_in, const int* in_sizes, int n_in,
                              void* d_out, int out_size, void* d_ws, size_t ws_size,
                              hipStream_t stream) {
  const int*   sentence = (const int*)  d_in[0];
  const float* emb      = (const float*)d_in[1];
  const float* Wih_f    = (const float*)d_in[2];
  const float* Whh_f    = (const float*)d_in[3];
  const float* bih_f    = (const float*)d_in[4];
  const float* bhh_f    = (const float*)d_in[5];
  const float* Wih_b    = (const float*)d_in[6];
  const float* Whh_b    = (const float*)d_in[7];
  const float* bih_b    = (const float*)d_in[8];
  const float* bhh_b    = (const float*)d_in[9];
  const float* W_out    = (const float*)d_in[10];
  const float* b_out    = (const float*)d_in[11];
  const float* trans    = (const float*)d_in[12];
  float* out = (float*)d_out;

  // workspace carve-up (256B aligned)
  char* ws = (char*)d_ws;
  size_t off = 0;
  auto carve = [&](size_t bytes) {
    char* p = ws + off;
    off = (off + bytes + 255) & ~(size_t)255;
    return p;
  };
  unsigned short* xbf    = (unsigned short*)carve((size_t)SEQ * EMBD * 2);       //  8 MB
  unsigned short* wcat   = (unsigned short*)carve((size_t)NCAT * EMBD * 2);      //  4 MB
  unsigned short* whhT_f = (unsigned short*)carve((size_t)HDIR * G4H * 2);       //  2 MB
  unsigned short* whhT_b = (unsigned short*)carve((size_t)HDIR * G4H * 2);       //  2 MB
  float*          xproj  = (float*)carve((size_t)SEQ * NCAT * 4);                // 128 MB
  float*          h_f    = (float*)carve((size_t)SEQ * HDIR * 4);                // 16 MB
  float*          h_b    = (float*)carve((size_t)SEQ * HDIR * 4);                // 16 MB
  float*          feats  = (float*)carve((size_t)SEQ * NTAG * 4);
  int*            bp     = (int*)carve((size_t)SEQ * NTAG * 4);
  (void)ws_size; (void)in_sizes; (void)n_in; (void)out_size;

  // 1) precision conversion + layout prep
  k_gather_bf16   <<<(SEQ * EMBD) / 256, 256, 0, stream>>>(sentence, emb, xbf);
  k_cvt_wih       <<<(NCAT * EMBD) / 256, 256, 0, stream>>>(Wih_f, Wih_b, wcat);
  k_trans_whh_pack<<<(HDIR * G4H) / 256, 256, 0, stream>>>(Whh_f, whhT_f);
  k_trans_whh_pack<<<(HDIR * G4H) / 256, 256, 0, stream>>>(Whh_b, whhT_b);

  // 2) big hoisted GEMM via WMMA: 512 m-tiles * 64 n-strips = 32768 waves
  k_gemm_xproj<<<(512 * 64) / 8, 256, 0, stream>>>(xbf, wcat, bih_f, bih_b, xproj);

  // 3) sequential recurrence, both directions concurrently
  k_lstm_rec<<<2, 1024, 0, stream>>>(xproj, whhT_f, whhT_b, bhh_f, bhh_b, h_f, h_b);

  // 4) tag features
  k_feats<<<SEQ, 256, 0, stream>>>(h_f, h_b, W_out, b_out, feats);

  // 5) Viterbi decode
  k_viterbi<<<1, 32, 0, stream>>>(feats, trans, bp, out);
}
